// GraphTransformerGFN_50714973831898
// MI455X (gfx1250) — compile-verified, matrix operand order checked by reading, not run on hardware
//
#include <hip/hip_runtime.h>
#include <hip/hip_bf16.h>

// ---------------------------------------------------------------------------
// Problem constants (from the reference)
// ---------------------------------------------------------------------------
#define NPG   50
#define NB    1024
#define NN    (NB * NPG)        // 51200 nodes
#define NE    204800            // directed edges
#define DD    128               // hidden dim
#define HH    2                 // heads
#define HD    (HH * DD)         // 256
#define FFD   512               // 4*D
#define NAUGC (NN + NB)         // 52224 augmented nodes
#define E2C   (NE + 2 * NN)     // 307200 edges before self loops
#define EAUGC (E2C + NAUGC)     // 359424 augmented edges
#define XDIM  71
#define XPAD  96
#define EDIM  4
#define EPAD  32
#define GDIM  32
#define EPS_GEN 1e-7f
#define EPS_LN  1e-5f
#define SCALE_INV 0.08838834764831845f   // 1/sqrt(128)
#define LRELU_SLOPE 0.01f

typedef __attribute__((ext_vector_type(16))) _Float16 v16h;
typedef __attribute__((ext_vector_type(8)))  float    v8f;

// ---------------------------------------------------------------------------
// WMMA GEMM: C[M,Ncol] = act(A[M,K](f16,row-major) x W^T[Ncol,K](f16) + bias)
// Weights are pre-transposed to [Ncol, Kpad] so BOTH A and B fragments are two
// contiguous 8-half groups per lane -> b128 loads, no strided scalar loads.
// 256 threads = 8 waves; each wave owns a 16x32 C tile (2 WMMAs sharing the A
// fragment); block tile = 32 rows x 128 cols. K stepped by 32 via
// v_wmma_f32_16x16x32_f16 (fp32 accumulate). Layouts per CDNA5 ISA 7.12.2.
// ---------------------------------------------------------------------------
template <int ACT, int OUT16>
__global__ void gemm_wmma(const _Float16* __restrict__ A,
                          const _Float16* __restrict__ Wt,   // [Ncol, K]
                          const float* __restrict__ bias,
                          void* __restrict__ Cout,
                          int M, int Ncol, int K) {
  const int wave = threadIdx.x >> 5;
  const int lane = threadIdx.x & 31;
  const int hs   = lane >> 4;      // lane-half select
  const int lr   = lane & 15;

  const int wy = wave >> 2;        // 0..1
  const int wx = wave & 3;         // 0..3
  const int row0 = blockIdx.y * 32 + wy * 16;
  const int col0 = blockIdx.x * 128 + wx * 32;
  if (row0 >= M || col0 >= Ncol) return;

  v8f acc0 = {};
  v8f acc1 = {};

  const _Float16* arowb = A  + (size_t)(row0 + lr) * K;
  const _Float16* b0b   = Wt + (size_t)(col0 + lr) * K;        // first 16-col tile
  const _Float16* b1b   = Wt + (size_t)(col0 + 16 + lr) * K;   // second 16-col tile

  const int steps = K >> 5;        // K is always a multiple of 32 here
  for (int s = 0; s < steps; ++s) {
    const int k0 = (s << 5) + hs * 8;

    const _Float16* ap = arowb + k0;
    __builtin_prefetch(ap + 32, 0, 1);   // global_prefetch_b8 next K-chunk
    v16h af, bf0, bf1;
#pragma unroll
    for (int i = 0; i < 8; ++i) { af[i] = ap[i];  af[8 + i] = ap[16 + i]; }
    const _Float16* bp0 = b0b + k0;
    const _Float16* bp1 = b1b + k0;
#pragma unroll
    for (int i = 0; i < 8; ++i) { bf0[i] = bp0[i]; bf0[8 + i] = bp0[16 + i]; }
#pragma unroll
    for (int i = 0; i < 8; ++i) { bf1[i] = bp1[i]; bf1[8 + i] = bp1[16 + i]; }

    acc0 = __builtin_amdgcn_wmma_f32_16x16x32_f16(false, af, false, bf0,
                                                  (short)0, acc0, false, false);
    acc1 = __builtin_amdgcn_wmma_f32_16x16x32_f16(false, af, false, bf1,
                                                  (short)0, acc1, false, false);
  }

  // ---- epilogue: C layout — VGPR r: row (row0 + hs*8 + r), col (colX + lr) ----
  const int colA = col0 + lr;
  const int colB = col0 + 16 + lr;
  const float bv0 = bias ? bias[colA] : 0.f;
  const float bv1 = bias ? bias[colB] : 0.f;
#pragma unroll
  for (int r = 0; r < 8; ++r) {
    const int row = row0 + hs * 8 + r;
    float va = acc0[r] + bv0;
    float vb = acc1[r] + bv1;
    if (ACT == 1) { va = (va > 0.f) ? va : LRELU_SLOPE * va;
                    vb = (vb > 0.f) ? vb : LRELU_SLOPE * vb; }
    if (ACT == 2) { va = fmaxf(va, 0.f); vb = fmaxf(vb, 0.f); }
    if (OUT16) {
      ((_Float16*)Cout)[(size_t)row * Ncol + colA] = (_Float16)va;
      ((_Float16*)Cout)[(size_t)row * Ncol + colB] = (_Float16)vb;
    } else {
      ((float*)Cout)[(size_t)row * Ncol + colA] = va;
      ((float*)Cout)[(size_t)row * Ncol + colB] = vb;
    }
  }
}

// ---------------------------------------------------------------------------
// Small helper kernels
// ---------------------------------------------------------------------------
__global__ void k_padcvt(const float* __restrict__ src, _Float16* __restrict__ dst,
                         int rows, int cols, int colspad) {
  long t = (long)blockIdx.x * 256 + threadIdx.x;
  if (t >= (long)rows * colspad) return;
  int r = (int)(t / colspad), c = (int)(t % colspad);
  dst[t] = (c < cols) ? (_Float16)src[(size_t)r * cols + c] : (_Float16)0.f;
}

// weight convert + TRANSPOSE + K-pad: w[kin,kout] (f32) -> dst[kout,kpad] (f16)
__global__ void k_wcvt_t(const float* __restrict__ w, _Float16* __restrict__ dst,
                         int kin, int kpad, int kout) {
  long t = (long)blockIdx.x * 256 + threadIdx.x;
  if (t >= (long)kpad * kout) return;
  int c = (int)(t / kpad), r = (int)(t % kpad);   // dst[c, r]
  dst[t] = (r < kin) ? (_Float16)w[(size_t)r * kout + c] : (_Float16)0.f;
}

__global__ void k_cvt(const float* __restrict__ a, _Float16* __restrict__ o, long n) {
  long t = (long)blockIdx.x * 256 + threadIdx.x;
  if (t < n) o[t] = (_Float16)a[t];
}

__global__ void k_addcvt(const float* __restrict__ a, const float* __restrict__ b,
                         _Float16* __restrict__ o, long n) {
  long t = (long)blockIdx.x * 256 + threadIdx.x;
  if (t < n) o[t] = (_Float16)(a[t] + b[t]);
}

__global__ void k_add_inplace(float* __restrict__ a, const float* __restrict__ b, long n) {
  long t = (long)blockIdx.x * 256 + threadIdx.x;
  if (t < n) a[t] += b[t];
}

__global__ void k_concat2cvt(const float* __restrict__ a, const float* __restrict__ b,
                             _Float16* __restrict__ h) {
  long t = (long)blockIdx.x * 256 + threadIdx.x;
  if (t >= (long)NAUGC * HD) return;
  int i = (int)(t >> 8), c = (int)(t & 255);
  float v = (c < DD) ? a[(size_t)i * DD + c] : b[(size_t)i * DD + (c - DD)];
  h[t] = (_Float16)v;
}

// augmented edge lists + augmented batch vector
__global__ void k_build_edges(const int* __restrict__ edge_index, const int* __restrict__ batch,
                              int* __restrict__ SRC, int* __restrict__ DST, int* __restrict__ AB) {
  long t = (long)blockIdx.x * 256 + threadIdx.x;
  if (t < NE)            { SRC[t] = edge_index[t]; DST[t] = edge_index[NE + t]; }
  else if (t < NE + NN)  { int i = (int)(t - NE);         SRC[t] = i;             DST[t] = batch[i] + NN; }
  else if (t < E2C)      { int i = (int)(t - NE - NN);    SRC[t] = batch[i] + NN; DST[t] = i; }
  else if (t < EAUGC)    { int i = (int)(t - E2C);        SRC[t] = i;             DST[t] = i; }
  if (t < NAUGC) AB[t] = (t < NN) ? (int)(t / NPG) : (int)(t - NN);
}

__global__ void k_count(const int* __restrict__ DST, float* __restrict__ cnt) {
  long t = (long)blockIdx.x * 256 + threadIdx.x;
  if (t < E2C) atomicAdd(&cnt[DST[t]], 1.f);
}

__global__ void k_loop_scatter(const float* __restrict__ Ebuf, const int* __restrict__ DST,
                               float* __restrict__ loop) {
  long t = (long)blockIdx.x * 256 + threadIdx.x;
  if (t >= (long)E2C * DD) return;
  int j = (int)(t >> 7), c = (int)(t & 127);
  float v = (j < NE) ? Ebuf[t] : ((c == 0) ? 1.f : 0.f);
  atomicAdd(&loop[(size_t)DST[j] * DD + c], v);
}

__global__ void k_loop_div(float* __restrict__ loop, const float* __restrict__ cnt) {
  long t = (long)blockIdx.x * 256 + threadIdx.x;
  if (t >= (long)NAUGC * DD) return;
  loop[t] /= fmaxf(cnt[t >> 7], 1.f);
}

__global__ void k_build_ae16(const float* __restrict__ Ebuf, const float* __restrict__ loop,
                             _Float16* __restrict__ ae16) {
  long t = (long)blockIdx.x * 256 + threadIdx.x;
  if (t >= (long)EAUGC * DD) return;
  int j = (int)(t >> 7), c = (int)(t & 127);
  float v;
  if (j < NE)       v = Ebuf[t];
  else if (j < E2C) v = (c == 0) ? 1.f : 0.f;
  else              v = loop[(size_t)(j - E2C) * DD + c];
  ae16[t] = (_Float16)v;
}

__global__ void k_c_embed(const float* __restrict__ cbuf, float* __restrict__ o,
                          _Float16* __restrict__ c16) {
  long t = (long)blockIdx.x * 256 + threadIdx.x;
  if (t >= (long)NB * DD) return;
  o[(size_t)NN * DD + t] = cbuf[t];
  c16[t] = (_Float16)cbuf[t];
}

// per-graph LayerNorm over (50 nodes + 1 virtual) x 128 channels
__global__ void k_graph_ln(const float* __restrict__ x, float* __restrict__ y) {
  const int g = blockIdx.x, tid = threadIdx.x;
  const int ROWS = NPG + 1;
  __shared__ float ssum[256], ssq[256];
  float s = 0.f, q = 0.f;
  for (int idx = tid; idx < ROWS * DD; idx += 256) {
    int r = idx / DD, c = idx % DD;
    int row = (r < NPG) ? (g * NPG + r) : (NN + g);
    float v = x[(size_t)row * DD + c];
    s += v; q += v * v;
  }
  ssum[tid] = s; ssq[tid] = q; __syncthreads();
  for (int st = 128; st > 0; st >>= 1) {
    if (tid < st) { ssum[tid] += ssum[tid + st]; ssq[tid] += ssq[tid + st]; }
    __syncthreads();
  }
  const float norm = (float)(ROWS * DD);
  const float mean = ssum[0] / norm;
  const float var  = ssq[0] / norm - mean * mean;
  const float inv  = rsqrtf(var + EPS_LN);
  for (int idx = tid; idx < ROWS * DD; idx += 256) {
    int r = idx / DD, c = idx % DD;
    int row = (r < NPG) ? (g * NPG + r) : (NN + g);
    y[(size_t)row * DD + c] = (x[(size_t)row * DD + c] - mean) * inv;
  }
}

// GENConv message scatter: agg[dst] += relu(o_n[src] + ae) + eps
__global__ void k_gen_msg(const float* __restrict__ on, const _Float16* __restrict__ ae16,
                          const int* __restrict__ SRC, const int* __restrict__ DST,
                          float* __restrict__ agg) {
  long t = (long)blockIdx.x * 256 + threadIdx.x;
  if (t >= (long)EAUGC * DD) return;
  int j = (int)(t >> 7), c = (int)(t & 127);
  float v = on[(size_t)SRC[j] * DD + c] + (float)ae16[t];
  v = fmaxf(v, 0.f) + EPS_GEN;
  atomicAdd(&agg[(size_t)DST[j] * DD + c], v);
}

// monotonic float<->uint encoding for atomicMax on floats
__device__ __forceinline__ unsigned encf(float f) {
  unsigned u = __float_as_uint(f);
  return (u & 0x80000000u) ? ~u : (u | 0x80000000u);
}
__device__ __forceinline__ float decf(unsigned e) {
  unsigned u = (e & 0x80000000u) ? (e & 0x7fffffffu) : ~e;
  return __uint_as_float(u);
}

// attention logits: one wave per edge, both heads; wave32 shfl reduction
__global__ void k_logits(const _Float16* __restrict__ q16, const _Float16* __restrict__ k16,
                         const _Float16* __restrict__ et16, const int* __restrict__ SRC,
                         const int* __restrict__ DST, float* __restrict__ alpha,
                         unsigned int* __restrict__ amax) {
  const int wave = threadIdx.x >> 5, lane = threadIdx.x & 31;
  long j = (long)blockIdx.x * 8 + wave;
  if (j >= EAUGC) return;
  const int s = SRC[j], d = DST[j];
  float a0 = 0.f, a1 = 0.f;
#pragma unroll
  for (int t = 0; t < 4; ++t) {
    int c = lane * 4 + t;
    a0 += (float)q16[(size_t)d * HD + c] *
          ((float)k16[(size_t)s * HD + c] + (float)et16[(size_t)j * HD + c]);
    a1 += (float)q16[(size_t)d * HD + DD + c] *
          ((float)k16[(size_t)s * HD + DD + c] + (float)et16[(size_t)j * HD + DD + c]);
  }
  for (int off = 16; off > 0; off >>= 1) {
    a0 += __shfl_down(a0, off, 32);
    a1 += __shfl_down(a1, off, 32);
  }
  if (lane == 0) {
    float l0 = a0 * SCALE_INV, l1 = a1 * SCALE_INV;
    alpha[j * 2]     = l0;
    alpha[j * 2 + 1] = l1;
    atomicMax(&amax[d * 2],     encf(l0));
    atomicMax(&amax[d * 2 + 1], encf(l1));
  }
}

__global__ void k_exp(const int* __restrict__ DST, const unsigned int* __restrict__ amax,
                      float* __restrict__ alpha, float* __restrict__ expsum) {
  long t = (long)blockIdx.x * 256 + threadIdx.x;
  if (t >= (long)EAUGC * 2) return;
  int j = (int)(t >> 1), h = (int)(t & 1);
  float m = decf(amax[DST[j] * 2 + h]);
  float ex = expf(alpha[t] - m);
  alpha[t] = ex;
  atomicAdd(&expsum[DST[j] * 2 + h], ex);
}

__global__ void k_attn_scatter(const _Float16* __restrict__ v16, const _Float16* __restrict__ et16,
                               const int* __restrict__ SRC, const int* __restrict__ DST,
                               const float* __restrict__ alpha, const float* __restrict__ expsum,
                               float* __restrict__ out) {
  long t = (long)blockIdx.x * 256 + threadIdx.x;
  if (t >= (long)EAUGC * HD) return;
  int j = (int)(t >> 8), c = (int)(t & 255), h = c >> 7;
  int d = DST[j], s = SRC[j];
  float a = alpha[(size_t)j * 2 + h] / (expsum[d * 2 + h] + 1e-16f);
  float v = ((float)v16[(size_t)s * HD + c] + (float)et16[t]) * a;
  atomicAdd(&out[(size_t)d * HD + c], v);
}

// FiLM: o += l_h * cs[:, :D] + cs[:, D:]
__global__ void k_film(float* __restrict__ o, const float* __restrict__ lh,
                       const float* __restrict__ cs, const int* __restrict__ AB) {
  long t = (long)blockIdx.x * 256 + threadIdx.x;
  if (t >= (long)NAUGC * DD) return;
  int i = (int)(t >> 7), c = (int)(t & 127);
  int g = AB[i];
  o[t] += lh[t] * cs[(size_t)g * HD + c] + cs[(size_t)g * HD + DD + c];
}

// global readout: glob[g] = [mean over 50 nodes, virtual node state]
__global__ void k_glob(const float* __restrict__ o, float* __restrict__ outg) {
  const int g = blockIdx.x, c = threadIdx.x;   // 128 threads
  float s = 0.f;
  for (int r = 0; r < NPG; ++r) s += o[(size_t)(g * NPG + r) * DD + c];
  outg[(size_t)g * (2 * DD) + c]      = s / (float)NPG;
  outg[(size_t)g * (2 * DD) + DD + c] = o[(size_t)(NN + g) * DD + c];
}

// ---------------------------------------------------------------------------
// Host side
// ---------------------------------------------------------------------------
static inline dim3 g1(long n) { return dim3((unsigned)((n + 255) / 256)); }

static void launch_gemm(const _Float16* A, const _Float16* Wt, const float* bias, void* C,
                        int M, int Ncol, int K, int act, int out16, hipStream_t s) {
  dim3 b(256), g(Ncol / 128, M / 32);
  if (out16) {
    if (act == 1)      gemm_wmma<1, 1><<<g, b, 0, s>>>(A, Wt, bias, C, M, Ncol, K);
    else if (act == 2) gemm_wmma<2, 1><<<g, b, 0, s>>>(A, Wt, bias, C, M, Ncol, K);
    else               gemm_wmma<0, 1><<<g, b, 0, s>>>(A, Wt, bias, C, M, Ncol, K);
  } else {
    if (act == 1)      gemm_wmma<1, 0><<<g, b, 0, s>>>(A, Wt, bias, C, M, Ncol, K);
    else if (act == 2) gemm_wmma<2, 0><<<g, b, 0, s>>>(A, Wt, bias, C, M, Ncol, K);
    else               gemm_wmma<0, 0><<<g, b, 0, s>>>(A, Wt, bias, C, M, Ncol, K);
  }
}

// param leaf indices (jax pytree flatten: dict keys sorted alphabetically)
// top-level params keys sorted: c2h(6), e2h(6), layers(3*19), x2h(6)
#define P_C2H 5
#define P_E2H 11
#define P_LAYER(l) (17 + 19 * (l))
#define P_X2H 74
// within a 3-linear MLP tuple: b0,w0,b1,w1,b2,w2
// within a layer dict (sorted): cscale.b=0 cscale.w=1 edge.w=2 ff0.b=3 ff0.w=4
//   ff1.b=5 ff1.w=6 gen.b=7 gen.w=8 k.b=9 k.w=10 lin.b=11 lin.w=12 q.b=13 q.w=14
//   skip.b=15 skip.w=16 v.b=17 v.w=18

extern "C" void kernel_launch(void* const* d_in, const int* in_sizes, int n_in,
                              void* d_out, int out_size, void* d_ws, size_t ws_size,
                              hipStream_t stream) {
  (void)in_sizes; (void)n_in; (void)out_size; (void)ws_size;

  const float* x_in   = (const float*)d_in[0];
  const float* ea_in  = (const float*)d_in[1];
  const float* cond   = (const float*)d_in[2];
  const int*   batch  = (const int*)d_in[3];
  const int*   eidx   = (const int*)d_in[4];
  auto P = [&](int i) { return (const float*)d_in[i]; };

  // ---- workspace carve-up ----
  char* wp = (char*)d_ws;
  auto alloc = [&](size_t bytes) -> void* {
    void* p = wp; wp += (bytes + 255) & ~(size_t)255; return p;
  };
  int*   SRC    = (int*)alloc((size_t)EAUGC * 4);
  int*   DST    = (int*)alloc((size_t)EAUGC * 4);
  int*   AB     = (int*)alloc((size_t)NAUGC * 4);
  unsigned int* amax = (unsigned int*)alloc((size_t)NAUGC * 2 * 4);
  float* o      = (float*)alloc((size_t)NAUGC * DD * 4);
  float* on     = (float*)alloc((size_t)NAUGC * DD * 4);
  float* agg    = (float*)alloc((size_t)NAUGC * DD * 4);   // also l_h / ff-out
  float* outb   = (float*)alloc((size_t)NAUGC * HD * 4);
  float* Ebuf   = (float*)alloc((size_t)NE * DD * 4);
  float* loop   = (float*)alloc((size_t)NAUGC * DD * 4);
  float* cbuf   = (float*)alloc((size_t)NB * DD * 4);
  float* cs     = (float*)alloc((size_t)NB * HD * 4);
  float* cnt    = (float*)alloc((size_t)NAUGC * 4);
  float* expsum = (float*)alloc((size_t)NAUGC * 2 * 4);
  float* alphab = (float*)alloc((size_t)EAUGC * 2 * 4);
  _Float16* ae16  = (_Float16*)alloc((size_t)EAUGC * DD * 2);
  _Float16* et16  = (_Float16*)alloc((size_t)EAUGC * HD * 2);
  _Float16* q16   = (_Float16*)alloc((size_t)NAUGC * HD * 2);
  _Float16* k16   = (_Float16*)alloc((size_t)NAUGC * HD * 2);
  _Float16* v16   = (_Float16*)alloc((size_t)NAUGC * HD * 2);
  _Float16* h16a  = (_Float16*)alloc((size_t)NE * DD * 2);      // generic f16 staging A
  _Float16* h16b  = (_Float16*)alloc((size_t)NAUGC * FFD * 2);  // generic f16 staging B
  _Float16* W16   = (_Float16*)alloc((size_t)262144 * 2);       // padded/transposed f16 weight
  _Float16* c16   = (_Float16*)alloc((size_t)NB * DD * 2);
  _Float16* c16in = (_Float16*)alloc((size_t)NB * GDIM * 2);

  // convert + transpose + pad weight: returns Wt [kout, kpad]
  auto wcvt = [&](const float* w, int kin, int kpad, int kout) -> const _Float16* {
    long n = (long)kpad * kout;
    k_wcvt_t<<<g1(n), 256, 0, stream>>>(w, W16, kin, kpad, kout);
    return W16;
  };

  // ================= Stage A: input MLPs (x2h, e2h, c2h) =================
  // x2h: [N,71]->128->128->128, leaky between layers
  k_padcvt<<<g1((long)NN * XPAD), 256, 0, stream>>>(x_in, h16a, NN, XDIM, XPAD);
  launch_gemm(h16a, wcvt(P(P_X2H + 1), XDIM, XPAD, DD), P(P_X2H + 0), h16b, NN, DD, XPAD, 1, 1, stream);
  launch_gemm(h16b, wcvt(P(P_X2H + 3), DD, DD, DD),     P(P_X2H + 2), h16a, NN, DD, DD,   1, 1, stream);
  launch_gemm(h16a, wcvt(P(P_X2H + 5), DD, DD, DD),     P(P_X2H + 4), o,    NN, DD, DD,   0, 0, stream);
  // e2h: [E,4]->128->128->128
  k_padcvt<<<g1((long)NE * EPAD), 256, 0, stream>>>(ea_in, h16a, NE, EDIM, EPAD);
  launch_gemm(h16a, wcvt(P(P_E2H + 1), EDIM, EPAD, DD), P(P_E2H + 0), h16b, NE, DD, EPAD, 1, 1, stream);
  launch_gemm(h16b, wcvt(P(P_E2H + 3), DD, DD, DD),     P(P_E2H + 2), h16a, NE, DD, DD,   1, 1, stream);
  launch_gemm(h16a, wcvt(P(P_E2H + 5), DD, DD, DD),     P(P_E2H + 4), Ebuf, NE, DD, DD,   0, 0, stream);
  // c2h: [B,32]->128->128->128
  k_padcvt<<<g1((long)NB * GDIM), 256, 0, stream>>>(cond, c16in, NB, GDIM, GDIM);
  launch_gemm(c16in, wcvt(P(P_C2H + 1), GDIM, GDIM, DD), P(P_C2H + 0), h16b, NB, DD, GDIM, 1, 1, stream);
  launch_gemm(h16b, wcvt(P(P_C2H + 3), DD, DD, DD),      P(P_C2H + 2), h16a, NB, DD, DD,   1, 1, stream);
  launch_gemm(h16a, wcvt(P(P_C2H + 5), DD, DD, DD),      P(P_C2H + 4), cbuf, NB, DD, DD,   0, 0, stream);
  k_c_embed<<<g1((long)NB * DD), 256, 0, stream>>>(cbuf, o, c16);  // o[N:] = c; c16 = f16(c)

  // ================= Stage B: edge augmentation + self-loop attrs =================
  k_build_edges<<<g1((long)EAUGC), 256, 0, stream>>>(eidx, batch, SRC, DST, AB);
  hipMemsetAsync(cnt,  0, (size_t)NAUGC * 4, stream);
  hipMemsetAsync(loop, 0, (size_t)NAUGC * DD * 4, stream);
  k_count<<<g1((long)E2C), 256, 0, stream>>>(DST, cnt);
  k_loop_scatter<<<g1((long)E2C * DD), 256, 0, stream>>>(Ebuf, DST, loop);
  k_loop_div<<<g1((long)NAUGC * DD), 256, 0, stream>>>(loop, cnt);
  k_build_ae16<<<g1((long)EAUGC * DD), 256, 0, stream>>>(Ebuf, loop, ae16);

  // ================= Stage C: L transformer layers =================
  for (int l = 0; l < 3; ++l) {
    const int PL = P_LAYER(l);
    // FiLM params: cs = cscale(c)
    launch_gemm(c16, wcvt(P(PL + 1), DD, DD, HD), P(PL + 0), cs, NB, HD, DD, 0, 0, stream);
    // pre-LN
    k_graph_ln<<<NB, 256, 0, stream>>>(o, on);
    // GENConv: agg = gen(o_n + sum_j relu(o_n[src]+ae)+eps)
    hipMemsetAsync(agg, 0, (size_t)NAUGC * DD * 4, stream);
    k_gen_msg<<<g1((long)EAUGC * DD), 256, 0, stream>>>(on, ae16, SRC, DST, agg);
    k_addcvt<<<g1((long)NAUGC * DD), 256, 0, stream>>>(on, agg, h16a, (long)NAUGC * DD);
    launch_gemm(h16a, wcvt(P(PL + 8), DD, DD, DD), P(PL + 7), agg, NAUGC, DD, DD, 0, 0, stream);
    // h = concat(o_n, agg) as f16
    k_concat2cvt<<<g1((long)NAUGC * HD), 256, 0, stream>>>(on, agg, h16b);
    // q,k,v (f16 outputs), skip (f32 into attention accumulator)
    launch_gemm(h16b, wcvt(P(PL + 14), HD, HD, HD), P(PL + 13), q16,  NAUGC, HD, HD, 0, 1, stream);
    launch_gemm(h16b, wcvt(P(PL + 10), HD, HD, HD), P(PL + 9),  k16,  NAUGC, HD, HD, 0, 1, stream);
    launch_gemm(h16b, wcvt(P(PL + 18), HD, HD, HD), P(PL + 17), v16,  NAUGC, HD, HD, 0, 1, stream);
    launch_gemm(h16b, wcvt(P(PL + 16), HD, HD, HD), P(PL + 15), outb, NAUGC, HD, HD, 0, 0, stream);
    // edge projection et = edge(ae), no bias, f16 out
    launch_gemm(ae16, wcvt(P(PL + 2), DD, DD, HD), nullptr, et16, EAUGC, HD, DD, 0, 1, stream);
    // segment softmax over dst (one wave per edge -> EAUG/8 blocks of 256)
    hipMemsetAsync(amax,   0, (size_t)NAUGC * 2 * 4, stream);   // enc(x) > 0 for all real x
    hipMemsetAsync(expsum, 0, (size_t)NAUGC * 2 * 4, stream);
    k_logits<<<dim3((EAUGC + 7) / 8), 256, 0, stream>>>(q16, k16, et16, SRC, DST, alphab, amax);
    k_exp<<<g1((long)EAUGC * 2), 256, 0, stream>>>(DST, amax, alphab, expsum);
    k_attn_scatter<<<g1((long)EAUGC * HD), 256, 0, stream>>>(v16, et16, SRC, DST, alphab, expsum, outb);
    // l_h = lin(out + skip)
    k_cvt<<<g1((long)NAUGC * HD), 256, 0, stream>>>(outb, h16a, (long)NAUGC * HD);
    launch_gemm(h16a, wcvt(P(PL + 12), HD, HD, DD), P(PL + 11), agg, NAUGC, DD, HD, 0, 0, stream);
    // FiLM residual
    k_film<<<g1((long)NAUGC * DD), 256, 0, stream>>>(o, agg, cs, AB);
    // feed-forward with pre-LN: o += ff1(leaky(ff0(ln(o))))
    k_graph_ln<<<NB, 256, 0, stream>>>(o, on);
    k_cvt<<<g1((long)NAUGC * DD), 256, 0, stream>>>(on, h16a, (long)NAUGC * DD);
    launch_gemm(h16a, wcvt(P(PL + 4), DD, DD, FFD), P(PL + 3), h16b, NAUGC, FFD, DD, 1, 1, stream);
    launch_gemm(h16b, wcvt(P(PL + 6), FFD, FFD, DD), P(PL + 5), agg, NAUGC, DD, FFD, 0, 0, stream);
    k_add_inplace<<<g1((long)NAUGC * DD), 256, 0, stream>>>(o, agg, (long)NAUGC * DD);
  }

  // ================= Stage D: outputs =================
  hipMemcpyAsync(d_out, o, (size_t)NN * DD * 4, hipMemcpyDeviceToDevice, stream);
  k_glob<<<NB, DD, 0, stream>>>(o, (float*)d_out + (size_t)NN * DD);
}
